// LinearCrossAttention_47467978555845
// MI455X (gfx1250) — matrix operand order repaired and checked by reading, not compiled, not run on hardware
//
#include <hip/hip_runtime.h>
#include <hip/hip_bf16.h>
#include <cstddef>
#include <cstdint>

// ---------------------------------------------------------------------------
// Types for CDNA5 WMMA bf16 path (wave32): v_wmma_f32_16x16x32_bf16
// ---------------------------------------------------------------------------
typedef __attribute__((ext_vector_type(16))) __bf16 v16bf;
typedef __attribute__((ext_vector_type(8)))  __bf16 v8bf;
typedef __attribute__((ext_vector_type(8)))  float  v8f;

union BF16x16 { v16bf v; v8bf h[2]; };

// A-fragment: 16x32 bf16 (MxK), ISA 7.12.2 layout.
__device__ __forceinline__ v16bf load_a_frag(const __bf16* __restrict__ A,
                                             int lda, int m0, int k0, int lane) {
  int m    = m0 + (lane & 15);
  int ksel = ((lane >> 4) & 1) << 3;  // 0 or 8
  const __bf16* p = A + (size_t)m * lda + (k0 + ksel);
  BF16x16 f;
  f.h[0] = *reinterpret_cast<const v8bf*>(p);
  f.h[1] = *reinterpret_cast<const v8bf*>(p + 16);
  return f.v;
}

// B-fragment: 32x16 bf16 (KxN). Lane L holds column N=n0+L%16; K = 8h+s (s<8), 16+8h+s-8 (s>=8).
__device__ __forceinline__ v16bf load_b_frag(const __bf16* __restrict__ Bt,
                                             int ldb, int k0, int n0, int lane) {
  const __bf16* p = Bt + (size_t)(k0 + (((lane >> 4) & 1) << 3)) * ldb + n0 + (lane & 15);
  BF16x16 f;
#pragma unroll
  for (int s = 0; s < 8; ++s) {
    f.v[s] = p[(size_t)s * ldb];
    f.v[8 + s] = p[(size_t)(16 + s) * ldb];
  }
  return f.v;
}

__device__ __forceinline__ v8f wmma_bf16(v16bf a, v16bf b, v8f c) {
  return __builtin_amdgcn_wmma_f32_16x16x32_bf16(
      false, a, false, b, (short)0, c, false, false);
}

// ---------------------------------------------------------------------------
// Conversion kernels
// ---------------------------------------------------------------------------
__global__ void cvt_f32_to_bf16(const float* __restrict__ in,
                                __bf16* __restrict__ out, size_t n) {
  size_t i = (size_t)blockIdx.x * blockDim.x + threadIdx.x;
  size_t stride = (size_t)gridDim.x * blockDim.x;
  for (; i < n; i += stride) out[i] = (__bf16)in[i];
}

// Wt[k*Dd + n] = (bf16) W[n*Dd + k]
__global__ void transpose_w_bf16(const float* __restrict__ W,
                                 __bf16* __restrict__ Wt, int Dd) {
  size_t i = (size_t)blockIdx.x * blockDim.x + threadIdx.x;
  size_t n = (size_t)Dd * Dd;
  size_t stride = (size_t)gridDim.x * blockDim.x;
  for (; i < n; i += stride) {
    int k = (int)(i / Dd);
    int c = (int)(i % Dd);
    Wt[i] = (__bf16)W[(size_t)c * Dd + k];
  }
}

// ---------------------------------------------------------------------------
// WMMA bf16 GEMM:  C[M,N] = A[M,K] @ Bt[K,N]; 4 waves, 128x128 block tile,
// 64x64 per wave, register double-buffered K loop (K % 64 == 0).
// EPI: 0 = bf16 store, 1 = elu+1 -> bf16, 2 = +bias -> f32,
//      3 = elu+1 -> bf16 stored per-head transposed: Kt[(b*H+h)][d][nk]
// ---------------------------------------------------------------------------
template <int EPI>
__global__ __launch_bounds__(128)
void gemm_wmma_bf16(const __bf16* __restrict__ A, const __bf16* __restrict__ Bt,
                    void* __restrict__ Cout, const float* __restrict__ bias,
                    int M, int N, int K, int NkP, int HP, int nkShift) {
  const int lane = threadIdx.x & 31;
  const int wave = threadIdx.x >> 5;
  const int m0 = blockIdx.y * 128 + (wave >> 1) * 64;
  const int n0 = blockIdx.x * 128 + (wave & 1) * 64;

  v8f c[4][4] = {};
  v16bf a0[4], b0[4], a1[4], b1[4];

#pragma unroll
  for (int i = 0; i < 4; ++i) a0[i] = load_a_frag(A, K, m0 + 16 * i, 0, lane);
#pragma unroll
  for (int j = 0; j < 4; ++j) b0[j] = load_b_frag(Bt, N, 0, n0 + 16 * j, lane);

  for (int k0 = 0; k0 < K; k0 += 64) {
#pragma unroll
    for (int i = 0; i < 4; ++i) a1[i] = load_a_frag(A, K, m0 + 16 * i, k0 + 32, lane);
#pragma unroll
    for (int j = 0; j < 4; ++j) b1[j] = load_b_frag(Bt, N, k0 + 32, n0 + 16 * j, lane);
#pragma unroll
    for (int i = 0; i < 4; ++i)
#pragma unroll
      for (int j = 0; j < 4; ++j) c[i][j] = wmma_bf16(a0[i], b0[j], c[i][j]);
    if (k0 + 64 < K) {
#pragma unroll
      for (int i = 0; i < 4; ++i) a0[i] = load_a_frag(A, K, m0 + 16 * i, k0 + 64, lane);
#pragma unroll
      for (int j = 0; j < 4; ++j) b0[j] = load_b_frag(Bt, N, k0 + 64, n0 + 16 * j, lane);
    }
#pragma unroll
    for (int i = 0; i < 4; ++i)
#pragma unroll
      for (int j = 0; j < 4; ++j) c[i][j] = wmma_bf16(a1[i], b1[j], c[i][j]);
  }

  // Epilogue: C frag element (r, lane): row = r + 8*(lane>=16), col = lane&15
#pragma unroll
  for (int i = 0; i < 4; ++i) {
    const int rb = m0 + i * 16 + (((lane >> 4) & 1) << 3);
#pragma unroll
    for (int j = 0; j < 4; ++j) {
      const int n = n0 + j * 16 + (lane & 15);
#pragma unroll
      for (int r = 0; r < 8; ++r) {
        const int row = rb + r;
        float v = c[i][j][r];
        if constexpr (EPI == 1 || EPI == 3) v = (v > 0.0f) ? (v + 1.0f) : __expf(v);
        if constexpr (EPI == 2) {
          ((float*)Cout)[(size_t)row * N + n] = v + bias[n];
        } else if constexpr (EPI == 3) {
          const int bb = row >> nkShift;
          const int nk = row & (NkP - 1);
          const int hh = n >> 6;
          const int dd = n & 63;
          ((__bf16*)Cout)[((size_t)(bb * HP + hh) * 64 + dd) * NkP + nk] = (__bf16)v;
        } else {
          ((__bf16*)Cout)[(size_t)row * N + n] = (__bf16)v;
        }
      }
    }
  }
}

// ---------------------------------------------------------------------------
// kv[b,h,d,e] = sum_n Kt[bh][d][n] * V[b,n,h*64+e]   (WMMA, K split over waves)
// ksum[b,h,d] = sum_n Kt[bh][d][n]
// One block (128 threads = 4 waves) per (b,h). Cross-wave reduce in LDS.
// ---------------------------------------------------------------------------
__global__ __launch_bounds__(128)
void kv_wmma_kernel(const __bf16* __restrict__ Kt,  // [B*H, 64, Nk]
                    const __bf16* __restrict__ V,   // [B*Nk, D]
                    __bf16* __restrict__ KV,        // [B*H, 64, 64]
                    float* __restrict__ KS,         // [B*H, 64]
                    int Nk, int D, int H) {
  const int bh = blockIdx.x;
  const int b = bh / H, h = bh % H;
  const __bf16* Ah = Kt + (size_t)bh * 64 * Nk;           // [64][Nk]
  const __bf16* Bh = V + (size_t)b * Nk * D + h * 64;     // rows nk, ld = D

  const int lane = threadIdx.x & 31;
  const int wave = threadIdx.x >> 5;
  const int kq = Nk >> 2;                 // per-wave K span
  const int kbeg = wave * kq;

  v8f c[4][4] = {};
  for (int k0 = kbeg; k0 < kbeg + kq; k0 += 32) {
    v16bf a[4], bf[4];
#pragma unroll
    for (int i = 0; i < 4; ++i) a[i] = load_a_frag(Ah, Nk, 16 * i, k0, lane);
#pragma unroll
    for (int j = 0; j < 4; ++j) bf[j] = load_b_frag(Bh, D, k0, 16 * j, lane);
#pragma unroll
    for (int i = 0; i < 4; ++i)
#pragma unroll
      for (int j = 0; j < 4; ++j) c[i][j] = wmma_bf16(a[i], bf[j], c[i][j]);
  }

  __shared__ float red[4][64][64];        // 64 KB
#pragma unroll
  for (int i = 0; i < 4; ++i) {
    const int rb = i * 16 + (((lane >> 4) & 1) << 3);
#pragma unroll
    for (int j = 0; j < 4; ++j) {
      const int col = j * 16 + (lane & 15);
#pragma unroll
      for (int r = 0; r < 8; ++r) red[wave][rb + r][col] = c[i][j][r];
    }
  }
  __syncthreads();

  __bf16* kvout = KV + (size_t)bh * 64 * 64;
  for (int idx = threadIdx.x; idx < 64 * 64; idx += 128) {
    const int rr = idx >> 6, cc = idx & 63;
    float s = red[0][rr][cc] + red[1][rr][cc] + red[2][rr][cc] + red[3][rr][cc];
    kvout[idx] = (__bf16)s;
  }

  // ksum: row sums of Kt (contiguous, vectorized)
  if (threadIdx.x < 64) {
    const v8bf* vr = reinterpret_cast<const v8bf*>(Ah + (size_t)threadIdx.x * Nk);
    float s = 0.0f;
    for (int i = 0; i < (Nk >> 3); ++i) {
      v8bf x = vr[i];
#pragma unroll
      for (int j = 0; j < 8; ++j) s += (float)x[j];
    }
    KS[(size_t)bh * 64 + threadIdx.x] = s;
  }
}

// ---------------------------------------------------------------------------
// attn[b,q,h*64+e] = z[q] * sum_d Q[b,q,h*64+d] * kv[b,h,d,e]
// kv tile staged to LDS via CDNA5 async global->LDS loads (ASYNCcnt path).
// ---------------------------------------------------------------------------
__global__ __launch_bounds__(128)
void attn_wmma_kernel(const __bf16* __restrict__ Q,   // [B*Nq, D]
                      const __bf16* __restrict__ KV,  // [B*H, 64, 64]
                      const float* __restrict__ KS,   // [B*H, 64]
                      __bf16* __restrict__ Attn,      // [B*Nq, D]
                      int Nq, int D, int H) {
  const int bh = blockIdx.y;
  const int b = bh / H, h = bh % H;
  const size_t rowBase = (size_t)b * Nq + (size_t)blockIdx.x * 64;
  const __bf16* Qh  = Q + rowBase * D + h * 64;
  const __bf16* kvh = KV + (size_t)bh * 64 * 64;
  const float*  ksh = KS + (size_t)bh * 64;

  __shared__ __align__(16) __bf16 kvs[64 * 64];
  __shared__ float zs[64];
  const int t = threadIdx.x;

  // --- async stage kv tile (8 KB) into LDS: 128 lanes x 4 x b128 ---
  {
    unsigned base = (unsigned)(size_t)(__attribute__((address_space(3))) void*)kvs;
#pragma unroll
    for (int it = 0; it < 4; ++it) {
      const int elem = (t + it * 128) * 8;           // 8 bf16 = 16 bytes
      unsigned lds_off = base + (unsigned)(elem * 2);
      unsigned long long gaddr = (unsigned long long)(size_t)(kvh + elem);
      asm volatile("global_load_async_to_lds_b128 %0, %1, off"
                   :: "v"(lds_off), "v"(gaddr) : "memory");
    }
    asm volatile("s_wait_asynccnt 0x0" ::: "memory");
  }

  if (t < 64) {
    const __bf16* qr = Qh + (size_t)t * D;
    float s = 0.0f;
#pragma unroll 8
    for (int d = 0; d < 64; ++d) s += (float)qr[d] * ksh[d];
    zs[t] = 1.0f / (s + 1e-6f);
  }
  __syncthreads();

  const int lane = t & 31;
  const int wave = t >> 5;
  const int mloc = wave * 16;

  v8f c[4] = {};
#pragma unroll
  for (int k0 = 0; k0 < 64; k0 += 32) {
    v16bf a = load_a_frag(Qh, D, mloc, k0, lane);
#pragma unroll
    for (int j = 0; j < 4; ++j) {
      // B-frag from LDS: lane holds col 16*j+lane%16, k in table order
      const __bf16* p = &kvs[(k0 + (((lane >> 4) & 1) << 3)) * 64 + 16 * j + (lane & 15)];
      BF16x16 f;
#pragma unroll
      for (int s = 0; s < 8; ++s) {
        f.v[s] = p[s * 64];
        f.v[8 + s] = p[(16 + s) * 64];
      }
      c[j] = wmma_bf16(a, f.v, c[j]);
    }
  }

  __bf16* outp = Attn + rowBase * D + h * 64;
  const int rb = mloc + (((lane >> 4) & 1) << 3);
#pragma unroll
  for (int j = 0; j < 4; ++j) {
    const int n = j * 16 + (lane & 15);
#pragma unroll
    for (int r = 0; r < 8; ++r) {
      const int row = rb + r;
      const float v = c[j][r] * zs[row];
      outp[(size_t)row * D + n] = (__bf16)v;
    }
  }
}

// ---------------------------------------------------------------------------
// Host launcher
// ---------------------------------------------------------------------------
extern "C" void kernel_launch(void* const* d_in, const int* in_sizes, int n_in,
                              void* d_out, int out_size, void* d_ws, size_t ws_size,
                              hipStream_t stream) {
  (void)in_sizes; (void)n_in; (void)out_size; (void)ws_size;
  const float* xq  = (const float*)d_in[0];
  const float* xkv = (const float*)d_in[1];
  const float* Wq  = (const float*)d_in[2];
  const float* Wk  = (const float*)d_in[3];
  const float* Wv  = (const float*)d_in[4];
  const float* Wo  = (const float*)d_in[5];
  const float* bo  = (const float*)d_in[6];
  float* out = (float*)d_out;

  const int B = 8, Nq = 4096, Nk = 1024, D = 768, H = 12;
  const int nkShift = 10;             // log2(Nk)
  const size_t Mq = (size_t)B * Nq;   // 32768
  const size_t Mk = (size_t)B * Nk;   // 8192

  char* ws = (char*)d_ws;
  size_t off = 0;
  auto alloc = [&](size_t bytes) -> char* {
    char* p = ws + off;
    off += (bytes + 255) & ~(size_t)255;
    return p;
  };

  __bf16* Xqb = (__bf16*)alloc(Mq * D * sizeof(__bf16));
  __bf16* Xkb = (__bf16*)alloc(Mk * D * sizeof(__bf16));
  __bf16* Wqt = (__bf16*)alloc((size_t)D * D * sizeof(__bf16));
  __bf16* Wkt = (__bf16*)alloc((size_t)D * D * sizeof(__bf16));
  __bf16* Wvt = (__bf16*)alloc((size_t)D * D * sizeof(__bf16));
  __bf16* Wot = (__bf16*)alloc((size_t)D * D * sizeof(__bf16));
  __bf16* Qb  = (__bf16*)alloc(Mq * D * sizeof(__bf16));
  __bf16* Ktb = (__bf16*)alloc((size_t)B * H * 64 * Nk * sizeof(__bf16));
  __bf16* Vb  = (__bf16*)alloc(Mk * D * sizeof(__bf16));
  __bf16* KVb = (__bf16*)alloc((size_t)B * H * 64 * 64 * sizeof(__bf16));
  float*  KSf = (float*) alloc((size_t)B * H * 64 * sizeof(float));
  __bf16* Atn = (__bf16*)alloc(Mq * D * sizeof(__bf16));

  // 1) conversions
  cvt_f32_to_bf16<<<8192, 256, 0, stream>>>(xq,  Xqb, Mq * D);
  cvt_f32_to_bf16<<<4096, 256, 0, stream>>>(xkv, Xkb, Mk * D);
  transpose_w_bf16<<<1152, 256, 0, stream>>>(Wq, Wqt, D);
  transpose_w_bf16<<<1152, 256, 0, stream>>>(Wk, Wkt, D);
  transpose_w_bf16<<<1152, 256, 0, stream>>>(Wv, Wvt, D);
  transpose_w_bf16<<<1152, 256, 0, stream>>>(Wo, Wot, D);

  // 2) projections (WMMA GEMMs)
  dim3 gq(D / 128, (unsigned)(Mq / 128));
  dim3 gk(D / 128, (unsigned)(Mk / 128));
  gemm_wmma_bf16<1><<<gq, 128, 0, stream>>>(Xqb, Wqt, Qb, nullptr, (int)Mq, D, D, 0, 0, 0);
  gemm_wmma_bf16<3><<<gk, 128, 0, stream>>>(Xkb, Wkt, Ktb, nullptr, (int)Mk, D, D, Nk, H, nkShift);
  gemm_wmma_bf16<0><<<gk, 128, 0, stream>>>(Xkb, Wvt, Vb, nullptr, (int)Mk, D, D, 0, 0, 0);

  // 3) kv + k_sum per head (WMMA)
  kv_wmma_kernel<<<B * H, 128, 0, stream>>>(Ktb, Vb, KVb, KSf, Nk, D, H);

  // 4) linear attention core (WMMA, async LDS staging)
  dim3 ga(Nq / 64, B * H);
  attn_wmma_kernel<<<ga, 128, 0, stream>>>(Qb, KVb, KSf, Atn, Nq, D, H);

  // 5) output projection + bias (f32 out)
  gemm_wmma_bf16<2><<<gq, 128, 0, stream>>>(Atn, Wot, out, bo, (int)Mq, D, D, 0, 0, 0);
}
